// FastSSM_81140522156901
// MI455X (gfx1250) — compile-verified, matrix operand order checked by reading, not run on hardware
//
#include <hip/hip_runtime.h>
#include <cstdint>
#include <cstddef>

// ---------------------------------------------------------------------------
// CDNA5 WMMA types (wave32): A/B = 16 bf16 (8 VGPRs), C/D = 8 f32.
// ---------------------------------------------------------------------------
typedef __attribute__((ext_vector_type(16))) __bf16 v16bf;
typedef __attribute__((ext_vector_type(8)))  float  v8f;

union Frag { uint4 u[2]; v16bf v; };

__device__ __forceinline__ unsigned short f2bf(float f) {
    union { float f; unsigned int u; } x; x.f = f;
    unsigned int r = 0x7FFFu + ((x.u >> 16) & 1u);   // round-to-nearest-even
    return (unsigned short)((x.u + r) >> 16);
}
__device__ __forceinline__ float bf2f(unsigned short b) {
    union { unsigned int u; float f; } x; x.u = ((unsigned int)b) << 16;
    return x.f;
}
__device__ __forceinline__ unsigned int pk2bf(float a, float b) {
    return (unsigned int)f2bf(a) | ((unsigned int)f2bf(b) << 16);
}
__device__ __forceinline__ void ldfrag(Frag& f, const unsigned short* p) {
    f.u[0] = *(const uint4*)p;
    f.u[1] = *(const uint4*)(p + 16);
}

// CDNA5 async memory->LDS copy, 16B per lane, no VGPR data (ASYNCcnt).
// GVS addressing: mem = SGPR_base + VGPR_offset + inst_offset;
// LDS  addr      = VDST(lane)    + inst_offset.
#define ASYNC_CP16(ldsoff, voff, base, imm)                                   \
    asm volatile("global_load_async_to_lds_b128 %0, %1, %2 offset:" #imm     \
                 :: "v"(ldsoff), "v"(voff), "s"(base) : "memory")
#define WAIT_ASYNC0() asm volatile("s_wait_asynccnt 0" ::: "memory")

// ---------------------------------------------------------------------------
// Pre-pass 1: f32 -> bf16 bulk convert (8 elements / thread, b128 in/out).
// ---------------------------------------------------------------------------
__global__ __launch_bounds__(256)
void convert_bf16_kernel(const float* __restrict__ src,
                         unsigned short* __restrict__ dst)
{
    const size_t i = ((size_t)blockIdx.x * 256 + threadIdx.x) * 8;
    const float4* s = (const float4*)(src + i);
    float4 a = s[0], b = s[1];
    uint4 o;
    o.x = pk2bf(a.x, a.y);  o.y = pk2bf(a.z, a.w);
    o.z = pk2bf(b.x, b.y);  o.w = pk2bf(b.z, b.w);
    *(uint4*)(dst + i) = o;
}

// ---------------------------------------------------------------------------
// Pre-pass 2: transpose+convert weights: W[K][N] f32 -> Wt[N][K] bf16 (once).
// ---------------------------------------------------------------------------
__global__ __launch_bounds__(256)
void transpose_bf16_kernel(const float* __restrict__ W,
                           unsigned short* __restrict__ Wt, int K, int N)
{
    __shared__ float tile[32][33];
    const int kb = blockIdx.x * 32, nb = blockIdx.y * 32;
    const int tx = threadIdx.x & 31, ty = threadIdx.x >> 5;   // 32 x 8
#pragma unroll
    for (int j = 0; j < 4; ++j)
        tile[ty + j * 8][tx] = W[(size_t)(kb + ty + j * 8) * N + nb + tx];
    __syncthreads();
#pragma unroll
    for (int j = 0; j < 4; ++j)
        Wt[(size_t)(nb + ty + j * 8) * K + kb + tx] = f2bf(tile[tx][ty + j * 8]);
}

// ---------------------------------------------------------------------------
// bf16 WMMA GEMM:  C[M,N] = act(A[M,K] * Bt[N,K]^T + bias[N])
// Tiles staged via global_load_async_to_lds_b128 into DOUBLE-BUFFERED LDS:
// zero staging VGPRs, next tile fetch overlaps current tile's 8 WMMAs,
// one s_wait_asynccnt + one barrier per K-chunk.
// Block: 256 threads = 8 wave32; tile 128(M) x 64(N); LDS K-chunk 64.
// ---------------------------------------------------------------------------
#define MT 128
#define NT 64
#define KT 64
#define KP (KT + 8)   // 144B rows: 16B-aligned chunks, bank-spread

template<bool OUT_SILU16>
__global__ __launch_bounds__(256)
void gemm_bf16_wmma(const unsigned short* __restrict__ A,
                    const unsigned short* __restrict__ Bt,
                    const float* __restrict__ bias,
                    void* __restrict__ Cout, int M, int N, int K)
{
    __shared__ unsigned short As[2][MT][KP];   // 2 x 18 KB
    __shared__ unsigned short Bs[2][NT][KP];   // 2 x  9 KB

    const int tid  = threadIdx.x;
    const int lane = tid & 31;
    const int wid  = tid >> 5;
    const int half = lane >> 4;     // 0: lanes 0-15, 1: lanes 16-31
    const int l16  = lane & 15;

    const int blockM = blockIdx.x * MT;
    const int blockN = blockIdx.y * NT;
    const int wm = (wid >> 1) * 32;          // 0..96
    const int wn = (wid & 1) * 32;           // 0 or 32

    // staging assignment (16B chunk granularity): A 4 chunks, B 2 chunks
    const int arow = tid >> 1, acol = (tid & 1) * 32;   // 128 rows x 64 k
    const int brow = tid >> 2, bcol = (tid & 3) * 16;   //  64 rows x 64 k
    const unsigned aGoff = (unsigned)(((size_t)(blockM + arow) * K + acol) * 2);
    const unsigned bGoff = (unsigned)(((size_t)(blockN + brow) * K + bcol) * 2);

    auto issue_tile = [&](int buf, int k0) {
        const unsigned la = (unsigned)(uintptr_t)&As[buf][arow][acol];
        const unsigned lb = (unsigned)(uintptr_t)&Bs[buf][brow][bcol];
        const unsigned ga = aGoff + (unsigned)(k0 * 2);
        const unsigned gb = bGoff + (unsigned)(k0 * 2);
        ASYNC_CP16(la, ga, A,  0);
        ASYNC_CP16(la, ga, A, 16);
        ASYNC_CP16(la, ga, A, 32);
        ASYNC_CP16(la, ga, A, 48);
        ASYNC_CP16(lb, gb, Bt, 0);
        ASYNC_CP16(lb, gb, Bt, 16);
    };

    v8f acc[4] = {};

    issue_tile(0, 0);                        // prologue fetch
    const int nk = K / KT;
    for (int kk = 0; kk < nk; ++kk) {
        const int cur = kk & 1;
        WAIT_ASYNC0();                       // my async writes to 'cur' landed
        __syncthreads();                     // everyone's landed; 'nxt' readers done
        if (kk + 1 < nk) issue_tile(cur ^ 1, (kk + 1) * KT);   // overlaps WMMAs

        // 2 K-subchunks x (2x2) 16x16 WMMAs from LDS buffer 'cur'
#pragma unroll
        for (int ks = 0; ks < 2; ++ks) {
            const int kc = ks * 32 + half * 8;
            Frag a0, a1, b0, b1;
            ldfrag(a0, &As[cur][wm + l16][kc]);
            ldfrag(a1, &As[cur][wm + 16 + l16][kc]);
            ldfrag(b0, &Bs[cur][wn + l16][kc]);
            ldfrag(b1, &Bs[cur][wn + 16 + l16][kc]);
            acc[0] = __builtin_amdgcn_wmma_f32_16x16x32_bf16(false, a0.v, false, b0.v,
                                                             (short)0, acc[0], false, false);
            acc[1] = __builtin_amdgcn_wmma_f32_16x16x32_bf16(false, a0.v, false, b1.v,
                                                             (short)0, acc[1], false, false);
            acc[2] = __builtin_amdgcn_wmma_f32_16x16x32_bf16(false, a1.v, false, b0.v,
                                                             (short)0, acc[2], false, false);
            acc[3] = __builtin_amdgcn_wmma_f32_16x16x32_bf16(false, a1.v, false, b1.v,
                                                             (short)0, acc[3], false, false);
        }
    }

    // epilogue: C/D layout row = vg + 8*half, col = l16
#pragma unroll
    for (int ti = 0; ti < 4; ++ti) {
        const int mi  = (ti >> 1) * 16;
        const int ni  = (ti & 1) * 16;
        const int col = blockN + wn + ni + l16;
        const float bv = bias[col];
#pragma unroll
        for (int vg = 0; vg < 8; ++vg) {
            const int row = blockM + wm + mi + vg + 8 * half;
            float v = acc[ti][vg] + bv;
            if (OUT_SILU16) {
                float s = v / (1.0f + __expf(-v));           // SiLU
                ((unsigned short*)Cout)[(size_t)row * N + col] = f2bf(s);
            } else {
                ((float*)Cout)[(size_t)row * N + col] = v;
            }
        }
    }
}

// ---------------------------------------------------------------------------
// SSM scan: thread per (batch, channel); 16 h-states in VGPRs; in-place
// bf16 read/write (L2-resident); pipelined next-xt load + global prefetch.
// ---------------------------------------------------------------------------
__global__ __launch_bounds__(256)
void ssm_scan(unsigned short* __restrict__ xy,
              const float* __restrict__ Bp, const float* __restrict__ Cpm,
              const float* __restrict__ Dp, int Bsz, int L, int d_inner)
{
    const int idx = blockIdx.x * blockDim.x + threadIdx.x;
    if (idx >= Bsz * d_inner) return;
    const int b = idx / d_inner;
    const int i = idx - b * d_inner;

    float bpr[16], cpr[16], h[16];
#pragma unroll
    for (int s = 0; s < 16; ++s) {
        bpr[s] = Bp[(size_t)i * 16 + s];           // Bp: [d_inner, 16]
        cpr[s] = Cpm[(size_t)s * d_inner + i];     // Cp: [16, d_inner]
        h[s]   = 0.0f;
    }
    const float dpr = Dp[i];

    unsigned short* p = xy + (size_t)b * L * d_inner + i;
    float xt = bf2f(*p);
    for (int t = 0; t < L; ++t) {
        __builtin_prefetch(p + (size_t)16 * d_inner, 0, 1);
        float xn = 0.0f;
        if (t + 1 < L) xn = bf2f(p[d_inner]);      // issued before the chain
        float y = dpr * xt;
#pragma unroll
        for (int s = 0; s < 16; ++s) {
            h[s] = h[s] * 0.9f + xt * bpr[s];
            y   += h[s] * cpr[s];
        }
        *p = f2bf(y);
        p += d_inner;
        xt = xn;
    }
}

// ---------------------------------------------------------------------------
// Launcher. Sizes fixed by reference: B=4, L=2048, d_model=1024,
// d_inner=2048, d_state=16, M = B*L = 8192. Workspace = 56 MB.
// ---------------------------------------------------------------------------
extern "C" void kernel_launch(void* const* d_in, const int* in_sizes, int n_in,
                              void* d_out, int out_size, void* d_ws, size_t ws_size,
                              hipStream_t stream)
{
    (void)in_sizes; (void)n_in; (void)out_size; (void)ws_size;

    const float* x    = (const float*)d_in[0];   // [4, 2048, 1024]
    const float* Win  = (const float*)d_in[1];   // [1024, 2048]
    const float* bin_ = (const float*)d_in[2];   // [2048]
    const float* Bp   = (const float*)d_in[3];   // [2048, 16]
    const float* Cp   = (const float*)d_in[4];   // [16, 2048]
    const float* Dp   = (const float*)d_in[5];   // [2048]
    const float* Wout = (const float*)d_in[6];   // [2048, 1024]
    const float* bout = (const float*)d_in[7];   // [1024]
    float* out = (float*)d_out;                  // [4, 2048, 1024]

    constexpr int Bsz = 4, L = 2048, d_model = 1024, d_inner = 2048;
    constexpr int M = Bsz * L;                   // 8192

    unsigned short* xa    = (unsigned short*)d_ws;                 // bf16 x,   16 MB
    unsigned short* xi    = xa   + (size_t)M * d_model;            // bf16 xi/y, 32 MB
    unsigned short* wint  = xi   + (size_t)M * d_inner;            // Win^T bf16, 4 MB
    unsigned short* woutt = wint + (size_t)d_inner * d_model;      // Wout^T bf16, 4 MB

    // Pre-pass: convert x; transpose+convert weights (once).
    convert_bf16_kernel<<<(size_t)M * d_model / 2048, 256, 0, stream>>>(x, xa);
    transpose_bf16_kernel<<<dim3(d_model / 32, d_inner / 32), 256, 0, stream>>>(
        Win, wint, d_model, d_inner);
    transpose_bf16_kernel<<<dim3(d_inner / 32, d_model / 32), 256, 0, stream>>>(
        Wout, woutt, d_inner, d_model);

    // Stage 1: xi = silu(x @ Win + bin)  -> xi (bf16)
    gemm_bf16_wmma<true><<<dim3(M / MT, d_inner / NT), 256, 0, stream>>>(
        xa, wint, bin_, (void*)xi, M, d_inner, d_model);

    // Stage 2: sequential scan over L, in place on xi
    ssm_scan<<<(Bsz * d_inner + 255) / 256, 256, 0, stream>>>(
        xi, Bp, Cp, Dp, Bsz, L, d_inner);

    // Stage 3: out = y @ Wout + bout  -> d_out (f32)
    gemm_bf16_wmma<false><<<dim3(M / MT, d_model / NT), 256, 0, stream>>>(
        xi, woutt, bout, (void*)out, M, d_model, d_inner);
}
